// SelfAttention_27384711479861
// MI455X (gfx1250) — matrix operand order repaired and checked by reading, not compiled, not run on hardware
//
#include <hip/hip_runtime.h>

typedef __attribute__((ext_vector_type(16))) _Float16 v16h;
typedef __attribute__((ext_vector_type(8)))  _Float16 v8h;
typedef __attribute__((ext_vector_type(4)))  _Float16 v4h;
typedef __attribute__((ext_vector_type(8)))  float    v8f;
typedef __attribute__((ext_vector_type(4)))  float    v4f;

#define C_DIM 256
#define N_TOK 4096
#define B_SZ  4
#define LDSROW 264   // x-tile row stride: 528B = 16B-aligned, bank-stride 4
#define KROW   264   // K-tile row stride (256 + 8 halves)
#define VROW   72    // V-tile row stride (64 + 8 halves)
#define WELEM  65536 // 256*256 weights per matrix

__device__ __forceinline__ v16h pack16(v8h lo, v8h hi) {
    v16h r;
#pragma unroll
    for (int i = 0; i < 8; ++i) { r[i] = lo[i]; r[8 + i] = hi[i]; }
    return r;
}

// A fragment (16xK=32 f16): lane<16 holds M=lane, K={k0+0..7, k0+16..23};
// lane>=16 holds M=lane-16, K={k0+8..15, k0+24..31}. (ISA 7.12.2)
template <typename PT>
__device__ __forceinline__ v16h load_a16(PT base, int row0, int ld, int k0, int lane) {
    int m    = lane & 15;
    int koff = (lane < 16) ? 0 : 8;
    PT  p    = base + (row0 + m) * ld + k0 + koff;
    v8h lo = *(const v8h*)(p);
    v8h hi = *(const v8h*)(p + 16);
    return pack16(lo, hi);
}

// B fragment (K=32 x 16 f16): lane<16 holds N=lane, K={k0+0..7, k0+16..23};
// lane>=16 holds N=lane-16, K={k0+8..15, k0+24..31}. Rows of `base` are the N index.
template <typename PT>
__device__ __forceinline__ v16h load_b16(PT base, int row0, int ld, int k0, int lane) {
    int n  = lane & 15;
    int kk = k0 + ((lane < 16) ? 0 : 8);
    PT  p  = base + (row0 + n) * ld + kk;
    v8h lo = *(const v8h*)(p);
    v8h hi = *(const v8h*)(p + 16);
    return pack16(lo, hi);
}

__device__ __forceinline__ v8f wmma16(v16h a, v16h b, v8f c) {
    return __builtin_amdgcn_wmma_f32_16x16x32_f16(false, a, false, b, (short)0, c,
                                                  false, false);
}

// ---------------------------------------------------------------------------
// Kernel 0: one-shot fp32 -> f16 conversion of Wq/Wk/Wv/Wo.
// ---------------------------------------------------------------------------
__global__ void __launch_bounds__(256)
cvt_weights(const float* __restrict__ Wq, const float* __restrict__ Wk,
            const float* __restrict__ Wv, const float* __restrict__ Wo,
            _Float16* __restrict__ dst) {
    const int i = (blockIdx.x * 256 + threadIdx.x) * 4;
    const float* srcs[4] = {Wq, Wk, Wv, Wo};
#pragma unroll
    for (int m = 0; m < 4; ++m) {
        v4f t = *(const v4f*)(srcs[m] + i);
        v4h o;
#pragma unroll
        for (int j = 0; j < 4; ++j) o[j] = (_Float16)t[j];
        *(v4h*)(dst + m * WELEM + i) = o;
    }
}

// ---------------------------------------------------------------------------
// Kernel 1: QKV projection.  grid (N/16, B, 3), block 512 (16 waves).
// Q -> qb[b][n][c] f16 (pre-scaled 1/sqrt(C)), K -> kb[b][n][c], V -> vtb[b][c][n]
// ---------------------------------------------------------------------------
__global__ void __launch_bounds__(512)
qkv_proj(const float* __restrict__ x, const _Float16* __restrict__ wf16,
         const float* __restrict__ bq, const float* __restrict__ bk,
         const float* __restrict__ bv,
         _Float16* __restrict__ qb, _Float16* __restrict__ kb,
         _Float16* __restrict__ vtb) {
    const int n0  = blockIdx.x * 16;
    const int b   = blockIdx.y;
    const int ms  = blockIdx.z;
    const _Float16* W = wf16 + (size_t)ms * WELEM;
    const float* bias = (ms == 0) ? bq : (ms == 1) ? bk : bv;
    const int tid  = threadIdx.x;
    const int lane = tid & 31;
    const int wid  = tid >> 5;

    __shared__ __align__(16) _Float16 lt[16 * LDSROW];
    {   // stage x tile transposed: lt[m][c] = x[b][c][n0+m], coalesced over m
        const float* xb = x + (size_t)b * C_DIM * N_TOK;
        int m = tid & 15;
        int cbase = tid >> 4;  // 0..31
#pragma unroll
        for (int r = 0; r < 8; ++r) {
            int c = cbase + r * 32;
            lt[m * LDSROW + c] = (_Float16)xb[(size_t)c * N_TOK + n0 + m];
        }
    }
    __syncthreads();

    const int co0 = wid * 16;
    v8f acc = {0.f, 0.f, 0.f, 0.f, 0.f, 0.f, 0.f, 0.f};
#pragma unroll
    for (int cs = 0; cs < 8; ++cs) {
        v16h a  = load_a16((const _Float16*)lt, 0, LDSROW, cs * 32, lane);
        v16h bw = load_b16(W, co0, C_DIM, cs * 32, lane);
        acc = wmma16(a, bw, acc);
    }
    const float bsv = bias[co0 + (lane & 15)];
    const int h = lane >> 4;
    if (ms == 2) {  // V transposed: vtb[b][c][n]; VGPR index v -> consecutive n
        _Float16* dst = vtb + (size_t)b * C_DIM * N_TOK
                            + (size_t)(co0 + (lane & 15)) * N_TOK + n0 + h * 8;
        v8h pk;
#pragma unroll
        for (int v = 0; v < 8; ++v) pk[v] = (_Float16)(acc[v] + bsv);
        *(v8h*)dst = pk;
    } else {        // Q (scaled) / K: [b][n][c]
        const float s = (ms == 0) ? 0.0625f : 1.0f;  // 1/sqrt(256)
        _Float16* dst = (ms == 0 ? qb : kb) + (size_t)b * N_TOK * C_DIM;
#pragma unroll
        for (int v = 0; v < 8; ++v) {
            int m = v + 8 * h;
            dst[(size_t)(n0 + m) * C_DIM + co0 + (lane & 15)] =
                (_Float16)((acc[v] + bsv) * s);
        }
    }
}

// ---------------------------------------------------------------------------
// Kernel 2: flash attention, 64-key blocks, block-shared K/V tiles in LDS,
// TWO 16-query tiles per wave (each LDS fragment feeds 2 WMMAs).
// grid (N/128, B), block 128 (4 waves, 32 queries each).
// ---------------------------------------------------------------------------
__global__ void __launch_bounds__(128)
attn_kernel(const _Float16* __restrict__ qb, const _Float16* __restrict__ kb,
            const _Float16* __restrict__ vtb, _Float16* __restrict__ ab) {
    const int tid   = threadIdx.x;
    const int lane  = tid & 31;
    const int wid   = tid >> 5;
    const int qbase = (blockIdx.x * 4 + wid) * 32;   // two 16-row query tiles
    const int b     = blockIdx.y;
    const _Float16* Q  = qb  + (size_t)b * N_TOK * C_DIM;
    const _Float16* K  = kb  + (size_t)b * N_TOK * C_DIM;
    const _Float16* Vt = vtb + (size_t)b * C_DIM * N_TOK;

    __shared__ __align__(16) _Float16 klds[64 * KROW];      // [key][c]
    __shared__ __align__(16) _Float16 vlds[C_DIM * VROW];   // [c][key]
    __shared__ __align__(16) _Float16 psh[4][2 * 16 * 64];  // per-wave P tiles
    _Float16* pr = &psh[wid][0];

    v16h qa[2][8];
#pragma unroll
    for (int u = 0; u < 2; ++u)
#pragma unroll
        for (int cs = 0; cs < 8; ++cs)
            qa[u][cs] = load_a16(Q, qbase + u * 16, C_DIM, cs * 32, lane);

    v8f acc[2][16];
#pragma unroll
    for (int u = 0; u < 2; ++u)
#pragma unroll
        for (int g = 0; g < 16; ++g) acc[u][g] = {0.f,0.f,0.f,0.f,0.f,0.f,0.f,0.f};
    float rowmax[2][8], rowsum[2][8];
#pragma unroll
    for (int u = 0; u < 2; ++u)
#pragma unroll
        for (int v = 0; v < 8; ++v) { rowmax[u][v] = -1e30f; rowsum[u][v] = 0.f; }

    const int h   = lane >> 4;
    const int col = lane & 15;

    for (int j0 = 0; j0 < N_TOK; j0 += 64) {
        __syncthreads();  // previous tile fully consumed
        {   // stage K tile: 64 rows x 512B -> 16 passes of (4 rows x 32 chunks)
            const int kr = tid >> 5, kc = tid & 31;
#pragma unroll
            for (int p = 0; p < 16; ++p) {
                int r = p * 4 + kr;
                v8h d = *(const v8h*)(K + (size_t)(j0 + r) * C_DIM + kc * 8);
                *(v8h*)(klds + r * KROW + kc * 8) = d;
            }
            // stage V tile: 256 rows x 128B -> 16 passes of (16 rows x 8 chunks)
            const int vr = tid >> 3, vc = tid & 7;
#pragma unroll
            for (int p = 0; p < 16; ++p) {
                int r = p * 16 + vr;
                v8h d = *(const v8h*)(Vt + (size_t)r * N_TOK + j0 + vc * 8);
                *(v8h*)(vlds + r * VROW + vc * 8) = d;
            }
        }
        __syncthreads();
        if (j0 + 64 < N_TOK) {  // hint next tile into cache (global_prefetch_b8)
            __builtin_prefetch(K + (size_t)(j0 + 64 + (tid >> 1)) * C_DIM +
                                   (tid & 1) * 128, 0, 3);
            __builtin_prefetch(Vt + (size_t)(tid * 2) * N_TOK + j0 + 64, 0, 3);
        }

        // ---- scores: 4 key tiles x 2 query tiles, 64 WMMAs; kf reused 2x
        v8f s[2][4];
#pragma unroll
        for (int u = 0; u < 2; ++u)
#pragma unroll
            for (int t = 0; t < 4; ++t) s[u][t] = {0.f,0.f,0.f,0.f,0.f,0.f,0.f,0.f};
#pragma unroll
        for (int cs = 0; cs < 8; ++cs)
#pragma unroll
            for (int t = 0; t < 4; ++t) {
                v16h kf = load_b16((const _Float16*)klds, t * 16, KROW, cs * 32, lane);
                s[0][t] = wmma16(qa[0][cs], kf, s[0][t]);
                s[1][t] = wmma16(qa[1][cs], kf, s[1][t]);
            }

        // ---- online softmax across 64 scores/row, both query tiles
        float corr[2][8];
#pragma unroll
        for (int u = 0; u < 2; ++u) {
            _Float16* pu = pr + u * 1024;
#pragma unroll
            for (int v = 0; v < 8; ++v) {
                float t = fmaxf(fmaxf(s[u][0][v], s[u][1][v]),
                                fmaxf(s[u][2][v], s[u][3][v]));
#pragma unroll
                for (int off = 1; off < 16; off <<= 1)
                    t = fmaxf(t, __shfl_xor(t, off, 32));
                float nm = fmaxf(rowmax[u][v], t);
                corr[u][v] = __expf(rowmax[u][v] - nm);
                rowmax[u][v] = nm;
                float p0 = __expf(s[u][0][v] - nm);
                float p1 = __expf(s[u][1][v] - nm);
                float p2 = __expf(s[u][2][v] - nm);
                float p3 = __expf(s[u][3][v] - nm);
                float ls = (p0 + p1) + (p2 + p3);
#pragma unroll
                for (int off = 1; off < 16; off <<= 1)
                    ls += __shfl_xor(ls, off, 32);
                rowsum[u][v] = rowsum[u][v] * corr[u][v] + ls;
                int m = v + 8 * h;
                pu[m * 64 + col]      = (_Float16)p0;   // keys j0+col
                pu[m * 64 + col + 16] = (_Float16)p1;
                pu[m * 64 + col + 32] = (_Float16)p2;
                pu[m * 64 + col + 48] = (_Float16)p3;
            }
#pragma unroll
            for (int g = 0; g < 16; ++g)
#pragma unroll
                for (int v = 0; v < 8; ++v) acc[u][g][v] *= corr[u][v];
        }
        __asm__ volatile("" ::: "memory");  // LDS is in-order within a wave

        // ---- P @ V: 64 WMMAs; vf reused 2x
        v16h pa[2][2];
#pragma unroll
        for (int u = 0; u < 2; ++u) {
            pa[u][0] = load_a16((const _Float16*)(pr + u * 1024), 0, 64, 0,  lane);
            pa[u][1] = load_a16((const _Float16*)(pr + u * 1024), 0, 64, 32, lane);
        }
#pragma unroll
        for (int g = 0; g < 16; ++g) {
            v16h vf0 = load_b16((const _Float16*)vlds, g * 16, VROW, 0,  lane);
            acc[0][g] = wmma16(pa[0][0], vf0, acc[0][g]);
            acc[1][g] = wmma16(pa[1][0], vf0, acc[1][g]);
            v16h vf1 = load_b16((const _Float16*)vlds, g * 16, VROW, 32, lane);
            acc[0][g] = wmma16(pa[0][1], vf1, acc[0][g]);
            acc[1][g] = wmma16(pa[1][1], vf1, acc[1][g]);
        }
    }

    _Float16* Ab = ab + (size_t)b * N_TOK * C_DIM;
#pragma unroll
    for (int u = 0; u < 2; ++u) {
        float inv[8];
#pragma unroll
        for (int v = 0; v < 8; ++v) inv[v] = 1.0f / rowsum[u][v];
#pragma unroll
        for (int g = 0; g < 16; ++g)
#pragma unroll
            for (int v = 0; v < 8; ++v)
                Ab[(size_t)(qbase + u * 16 + v + 8 * h) * C_DIM + g * 16 + col] =
                    (_Float16)(acc[u][g][v] * inv[v]);
    }
}

// ---------------------------------------------------------------------------
// Kernel 3: output projection + residual.  grid (N/16, B), block 512 (16 waves)
// out[b][co][n] = bo[co] + x[b][co][n] + sum_c ab[b][n][c] * Wo[co][c]
// ---------------------------------------------------------------------------
__global__ void __launch_bounds__(512)
out_proj(const _Float16* __restrict__ ab, const _Float16* __restrict__ Woh,
         const float* __restrict__ bo, const float* __restrict__ x,
         float* __restrict__ out) {
    const int n0   = blockIdx.x * 16;
    const int b    = blockIdx.y;
    const int lane = threadIdx.x & 31;
    const int wid  = threadIdx.x >> 5;
    const int co0  = wid * 16;
    const _Float16* A = ab + (size_t)b * N_TOK * C_DIM;

    v8f acc = {0.f, 0.f, 0.f, 0.f, 0.f, 0.f, 0.f, 0.f};
#pragma unroll
    for (int cs = 0; cs < 8; ++cs) {
        v16h a = load_a16(A, n0, C_DIM, cs * 32, lane);
        v16h w = load_b16(Woh, co0, C_DIM, cs * 32, lane);
        acc = wmma16(a, w, acc);
    }
    const int h  = lane >> 4;
    const int co = co0 + (lane & 15);
    const float bias = bo[co];
    const size_t base = (size_t)b * C_DIM * N_TOK + (size_t)co * N_TOK + n0 + h * 8;
    v4f x0 = *(const v4f*)(x + base);
    v4f x1 = *(const v4f*)(x + base + 4);
    v4f o0, o1;
#pragma unroll
    for (int i = 0; i < 4; ++i) {
        o0[i] = acc[i]     + bias + x0[i];
        o1[i] = acc[4 + i] + bias + x1[i];
    }
    *(v4f*)(out + base)     = o0;
    *(v4f*)(out + base + 4) = o1;
}

// ---------------------------------------------------------------------------
extern "C" void kernel_launch(void* const* d_in, const int* in_sizes, int n_in,
                              void* d_out, int out_size, void* d_ws, size_t ws_size,
                              hipStream_t stream) {
    const float* x  = (const float*)d_in[0];
    const float* Wq = (const float*)d_in[1];
    const float* bq = (const float*)d_in[2];
    const float* Wk = (const float*)d_in[3];
    const float* bk = (const float*)d_in[4];
    const float* Wv = (const float*)d_in[5];
    const float* bv = (const float*)d_in[6];
    const float* Wo = (const float*)d_in[7];
    const float* bo = (const float*)d_in[8];
    float* out = (float*)d_out;

    _Float16* ws = (_Float16*)d_ws;
    const size_t per = (size_t)B_SZ * N_TOK * C_DIM;  // 4M halves = 8 MB each
    _Float16* qbuf = ws;
    _Float16* kbuf = ws + per;
    _Float16* vtbf = ws + 2 * per;
    _Float16* abuf = ws + 3 * per;
    _Float16* wf16 = ws + 4 * per;                    // 4 x 65536 halves

    cvt_weights<<<dim3(WELEM / 1024), 256, 0, stream>>>(Wq, Wk, Wv, Wo, wf16);
    qkv_proj<<<dim3(N_TOK / 16, B_SZ, 3), 512, 0, stream>>>(
        x, wf16, bq, bk, bv, qbuf, kbuf, vtbf);
    attn_kernel<<<dim3(N_TOK / 128, B_SZ), 128, 0, stream>>>(
        qbuf, kbuf, vtbf, abuf);
    out_proj<<<dim3(N_TOK / 16, B_SZ), 512, 0, stream>>>(
        abuf, wf16 + 3 * WELEM, bo, x, out);
}
